// RNNModel_266287972946
// MI455X (gfx1250) — compile-verified, matrix operand order checked by reading, not run on hardware
//
#include <hip/hip_runtime.h>
#include <hip/hip_bf16.h>
#include <stddef.h>

#ifndef __has_builtin
#define __has_builtin(x) 0
#endif
#if __has_builtin(__builtin_amdgcn_global_load_async_to_lds_b128) && \
    __has_builtin(__builtin_amdgcn_s_wait_asynccnt)
#define HAVE_ASYNC_LDS 1
#else
#define HAVE_ASYNC_LDS 0
#endif

// ---------------- types ----------------
typedef __attribute__((ext_vector_type(8)))  float  v8f;
typedef __attribute__((ext_vector_type(8)))  __bf16 v8bf;
typedef __attribute__((ext_vector_type(16))) __bf16 v16bf;
typedef __attribute__((ext_vector_type(4)))  int    v4i;

// float -> bf16, round-to-nearest-even (no libcalls)
static __device__ __forceinline__ __bf16 f2bf(float f) {
    unsigned u = __builtin_bit_cast(unsigned, f);
    unsigned r = (u + 0x7FFFu + ((u >> 16) & 1u)) >> 16;
    unsigned short s = (unsigned short)r;
    return __builtin_bit_cast(__bf16, s);
}

static __device__ __forceinline__ v16bf mk_frag(v8bf lo, v8bf hi) {
    v16bf r;
#pragma unroll
    for (int i = 0; i < 8; ++i) { r[i] = lo[i]; r[i + 8] = hi[i]; }
    return r;
}

static __device__ __forceinline__ float sigm(float x) {
    return 1.0f / (1.0f + __expf(-x));
}
static __device__ __forceinline__ float tanh_fast(float x) {
    float e = __expf(2.0f * x);
    return (e - 1.0f) / (e + 1.0f);
}

#if HAVE_ASYNC_LDS
// per-lane async copy of 16B global -> 16B LDS, tracked by ASYNCcnt
static __device__ __forceinline__ void async_cp16(const __bf16* g, __bf16* l) {
    __builtin_amdgcn_global_load_async_to_lds_b128(
        (__attribute__((address_space(1))) v4i*)g,
        (__attribute__((address_space(3))) v4i*)l, 0, 0);
}
#endif

// ---------------- setup / conversion kernels ----------------

// src f32 [N,K] row-major -> dst bf16 [NP,KP] zero-padded
__global__ void cvt_weight(const float* __restrict__ src, __bf16* __restrict__ dst,
                           int N, int K, int NP, int KP) {
    size_t i = (size_t)blockIdx.x * 256 + threadIdx.x;
    size_t tot = (size_t)NP * KP;
    if (i >= tot) return;
    int n = (int)(i / KP), k = (int)(i % KP);
    float v = (n < N && k < K) ? src[(size_t)n * K + k] : 0.0f;
    dst[i] = f2bf(v);
}

__global__ void bias_sum(const float* __restrict__ a, const float* __restrict__ b,
                         float* __restrict__ o, int N, int NP) {
    int i = blockIdx.x * 256 + threadIdx.x;
    if (i >= NP) return;
    o[i] = (i < N) ? (a[i] + b[i]) : 0.0f;
}

// tokens [S*B] -> emb bf16 [S*B, KP] (zero-padded beyond NINP=400)
__global__ void embed_kernel(const int* __restrict__ tok, const float* __restrict__ embW,
                             __bf16* __restrict__ out, int KP) {
    size_t i = (size_t)blockIdx.x * 256 + threadIdx.x;
    size_t tot = (size_t)25600 * KP;
    if (i >= tot) return;
    int row = (int)(i / KP), col = (int)(i % KP);
    float v = (col < 400) ? embW[(size_t)tok[row] * 400 + col] : 0.0f;
    out[i] = f2bf(v);
}

__global__ void zero_bf16(__bf16* __restrict__ p, size_t n) {
    size_t i = (size_t)blockIdx.x * 256 + threadIdx.x;
    if (i < n) p[i] = __builtin_bit_cast(__bf16, (unsigned short)0);
}

__global__ void copy_f32(float* __restrict__ dst, const float* __restrict__ src, size_t n) {
    size_t i = (size_t)blockIdx.x * 256 + threadIdx.x;
    if (i < n) dst[i] = src[i];
}

// h0/c0 f32 [128,H] -> hb bf16 [128,KP] (padded) and c f32 [128,H]
__global__ void init_state(const float* __restrict__ h0, const float* __restrict__ c0,
                           __bf16* __restrict__ hb, float* __restrict__ cst, int H, int KP) {
    int i = blockIdx.x * 256 + threadIdx.x;
    if (i >= 128 * KP) return;
    int b = i / KP, j = i % KP;
    hb[i] = f2bf(j < H ? h0[b * H + j] : 0.0f);
    if (j < H) cst[b * H + j] = c0[b * H + j];
}

// ---------------- WMMA GEMMs ----------------
// Out[M,NP] = A[M,KP](bf16) x W[NP,KP]^T(bf16) + bias[NP]   (NT GEMM)
// grid = (NP/64, M/128), block = 256 (8 waves); wave w -> M-tile; all waves
// share a 64x32 bf16 B-tile staged through LDS (async double-buffered when
// the toolchain exposes GLOBAL_LOAD_ASYNC_TO_LDS).
__global__ void wmma_gemm_bias(const __bf16* __restrict__ A, const __bf16* __restrict__ W,
                               const float* __restrict__ bias, float* __restrict__ Out,
                               int KP, int NP) {
#if HAVE_ASYNC_LDS
    __shared__ __bf16 sw[2][64 * 32];
#else
    __shared__ __bf16 sw[1][64 * 32];
#endif
    int tid  = threadIdx.x;
    int wave = tid >> 5, lane = tid & 31;
    int lm = lane & 15, half = lane >> 4;
    int ncol = blockIdx.x * 64;
    int mrow = (blockIdx.y * 8 + wave) * 16;

    v8f acc[4];
#pragma unroll
    for (int j = 0; j < 4; ++j) {
        float bv = bias[ncol + 16 * j + lm];
#pragma unroll
        for (int r = 0; r < 8; ++r) acc[j][r] = bv;
    }

    int srow = (tid * 8) >> 5;   // 0..63
    int scol = (tid * 8) & 31;   // 0,8,16,24

#if HAVE_ASYNC_LDS
    async_cp16(&W[(size_t)(ncol + srow) * KP + scol], &sw[0][srow * 32 + scol]);
    for (int k = 0; k < KP; k += 32) {
        int cur = (k >> 5) & 1;
        if (k + 32 < KP) {
            async_cp16(&W[(size_t)(ncol + srow) * KP + (k + 32) + scol],
                       &sw[cur ^ 1][srow * 32 + scol]);
            __builtin_amdgcn_s_wait_asynccnt(1);
        } else {
            __builtin_amdgcn_s_wait_asynccnt(0);
        }
        __syncthreads();
        const __bf16* ap = A + (size_t)(mrow + lm) * KP + k + half * 8;
        v16bf a = mk_frag(*(const v8bf*)ap, *(const v8bf*)(ap + 16));
#pragma unroll
        for (int j = 0; j < 4; ++j) {
            const __bf16* wp = &sw[cur][(16 * j + lm) * 32 + half * 8];
            v16bf bm = mk_frag(*(const v8bf*)wp, *(const v8bf*)(wp + 16));
            acc[j] = __builtin_amdgcn_wmma_f32_16x16x32_bf16(
                false, a, false, bm, (short)0, acc[j], false, false);
        }
        __syncthreads();
    }
#else
    for (int k = 0; k < KP; k += 32) {
        __syncthreads();
        *(v8bf*)&sw[0][srow * 32 + scol] =
            *(const v8bf*)&W[(size_t)(ncol + srow) * KP + k + scol];
        __syncthreads();
        const __bf16* ap = A + (size_t)(mrow + lm) * KP + k + half * 8;
        v16bf a = mk_frag(*(const v8bf*)ap, *(const v8bf*)(ap + 16));
#pragma unroll
        for (int j = 0; j < 4; ++j) {
            const __bf16* wp = &sw[0][(16 * j + lm) * 32 + half * 8];
            v16bf bm = mk_frag(*(const v8bf*)wp, *(const v8bf*)(wp + 16));
            acc[j] = __builtin_amdgcn_wmma_f32_16x16x32_bf16(
                false, a, false, bm, (short)0, acc[j], false, false);
        }
    }
#endif

    int m0 = mrow + half * 8;
#pragma unroll
    for (int j = 0; j < 4; ++j) {
        int nn = ncol + 16 * j + lm;
#pragma unroll
        for (int r = 0; r < 8; ++r)
            Out[(size_t)(m0 + r) * NP + nn] = acc[j][r];
    }
}

// g[128,NP] = xg_t[128,NP] + h[128,KP](bf16) x Whh[NP,KP]^T(bf16)
// grid = (NP/64), block = 256; wave w -> M-tile w (M=128 fixed). All 8 waves
// consume the SAME 64x32 Whh tile, so it is staged through LDS (async when
// available) instead of being fetched 8x from global.
__global__ void wmma_step_gemm(const __bf16* __restrict__ h, const __bf16* __restrict__ W,
                               const float* __restrict__ xg, float* __restrict__ g,
                               int KP, int NP) {
#if HAVE_ASYNC_LDS
    __shared__ __bf16 sw[2][64 * 32];
#else
    __shared__ __bf16 sw[1][64 * 32];
#endif
    int tid  = threadIdx.x;
    int wave = tid >> 5, lane = tid & 31;
    int lm = lane & 15, half = lane >> 4;
    int mrow = wave * 16;
    int ncol = blockIdx.x * 64;
    int m0 = mrow + half * 8;

    v8f acc[4];
#pragma unroll
    for (int j = 0; j < 4; ++j) {
        int nn = ncol + 16 * j + lm;
#pragma unroll
        for (int r = 0; r < 8; ++r) acc[j][r] = xg[(size_t)(m0 + r) * NP + nn];
    }

    int srow = (tid * 8) >> 5;   // 0..63
    int scol = (tid * 8) & 31;   // 0,8,16,24

#if HAVE_ASYNC_LDS
    async_cp16(&W[(size_t)(ncol + srow) * KP + scol], &sw[0][srow * 32 + scol]);
    for (int k = 0; k < KP; k += 32) {
        int cur = (k >> 5) & 1;
        if (k + 32 < KP) {
            async_cp16(&W[(size_t)(ncol + srow) * KP + (k + 32) + scol],
                       &sw[cur ^ 1][srow * 32 + scol]);
            __builtin_amdgcn_s_wait_asynccnt(1);
        } else {
            __builtin_amdgcn_s_wait_asynccnt(0);
        }
        __syncthreads();
        const __bf16* ap = h + (size_t)(mrow + lm) * KP + k + half * 8;
        v16bf a = mk_frag(*(const v8bf*)ap, *(const v8bf*)(ap + 16));
#pragma unroll
        for (int j = 0; j < 4; ++j) {
            const __bf16* wp = &sw[cur][(16 * j + lm) * 32 + half * 8];
            v16bf bm = mk_frag(*(const v8bf*)wp, *(const v8bf*)(wp + 16));
            acc[j] = __builtin_amdgcn_wmma_f32_16x16x32_bf16(
                false, a, false, bm, (short)0, acc[j], false, false);
        }
        __syncthreads();
    }
#else
    for (int k = 0; k < KP; k += 32) {
        __syncthreads();
        *(v8bf*)&sw[0][srow * 32 + scol] =
            *(const v8bf*)&W[(size_t)(ncol + srow) * KP + k + scol];
        __syncthreads();
        const __bf16* ap = h + (size_t)(mrow + lm) * KP + k + half * 8;
        v16bf a = mk_frag(*(const v8bf*)ap, *(const v8bf*)(ap + 16));
#pragma unroll
        for (int j = 0; j < 4; ++j) {
            const __bf16* wp = &sw[0][(16 * j + lm) * 32 + half * 8];
            v16bf bm = mk_frag(*(const v8bf*)wp, *(const v8bf*)(wp + 16));
            acc[j] = __builtin_amdgcn_wmma_f32_16x16x32_bf16(
                false, a, false, bm, (short)0, acc[j], false, false);
        }
    }
#endif

#pragma unroll
    for (int j = 0; j < 4; ++j) {
        int nn = ncol + 16 * j + lm;
#pragma unroll
        for (int r = 0; r < 8; ++r)
            g[(size_t)(m0 + r) * NP + nn] = acc[j][r];
    }
}

// gates + state update: c' = sig(f)*c + sig(i)*tanh(g); h' = sig(o)*tanh(c')
__global__ void lstm_gates(const float* __restrict__ g, float* __restrict__ cst,
                           float* __restrict__ hf, __bf16* __restrict__ hb,
                           __bf16* __restrict__ ybf, float* __restrict__ yf,
                           int H, int NP, int KP, int t) {
    int i = blockIdx.x * 256 + threadIdx.x;
    if (i >= 128 * H) return;
    int b = i / H, j = i % H;
    const float* gb = g + (size_t)b * NP;
    float ii = gb[j], ff = gb[H + j], gg = gb[2 * H + j], oo = gb[3 * H + j];
    float cn = sigm(ff) * cst[i] + sigm(ii) * tanh_fast(gg);
    float hn = sigm(oo) * tanh_fast(cn);
    cst[i] = cn;
    hf[i] = hn;
    hb[(size_t)b * KP + j] = f2bf(hn);
    size_t row = (size_t)t * 128 + b;
    if (ybf) ybf[row * KP + j] = f2bf(hn);
    if (yf)  yf[row * H + j] = hn;
}

// ---------------- host launch ----------------
extern "C" void kernel_launch(void* const* d_in, const int* in_sizes, int n_in,
                              void* d_out, int out_size, void* d_ws, size_t ws_size,
                              hipStream_t stream) {
    (void)in_sizes; (void)n_in; (void)out_size; (void)ws_size;

    const int*   tokens = (const int*)  d_in[0];
    const float* embW   = (const float*)d_in[1];
    const float* Wih0   = (const float*)d_in[2];
    const float* Whh0   = (const float*)d_in[3];
    const float* bih0   = (const float*)d_in[4];
    const float* bhh0   = (const float*)d_in[5];
    const float* Wih1   = (const float*)d_in[6];
    const float* Whh1   = (const float*)d_in[7];
    const float* bih1   = (const float*)d_in[8];
    const float* bhh1   = (const float*)d_in[9];
    const float* Wih2   = (const float*)d_in[10];
    const float* Whh2   = (const float*)d_in[11];
    const float* bih2   = (const float*)d_in[12];
    const float* bhh2   = (const float*)d_in[13];
    const float* h0_0   = (const float*)d_in[14];
    const float* c0_0   = (const float*)d_in[15];
    const float* h0_1   = (const float*)d_in[16];
    const float* c0_1   = (const float*)d_in[17];
    const float* h0_2   = (const float*)d_in[18];
    const float* c0_2   = (const float*)d_in[19];
    float* out = (float*)d_out;

    constexpr int    NINP = 400, NHID = 1150;
    constexpr int    KP0 = 416, KP1 = 1152, NP01 = 4608, NP2 = 1600;
    constexpr size_t M = 25600; // S*B

    // workspace sub-allocation (256B aligned)
    char* base = (char*)d_ws;
    size_t off = 0;
    auto alloc = [&](size_t bytes) -> void* {
        off = (off + 255) & ~(size_t)255;
        void* r = base + off;
        off += bytes;
        return r;
    };
    __bf16* embb  = (__bf16*)alloc(M * KP0 * 2);
    __bf16* wih0b = (__bf16*)alloc((size_t)NP01 * KP0 * 2);
    __bf16* whh0b = (__bf16*)alloc((size_t)NP01 * KP1 * 2);
    __bf16* wih1b = (__bf16*)alloc((size_t)NP01 * KP1 * 2);
    __bf16* whh1b = (__bf16*)alloc((size_t)NP01 * KP1 * 2);
    __bf16* wih2b = (__bf16*)alloc((size_t)NP2  * KP1 * 2);
    __bf16* whh2b = (__bf16*)alloc((size_t)NP2  * KP0 * 2);
    float*  bs0   = (float*) alloc((size_t)NP01 * 4);
    float*  bs1   = (float*) alloc((size_t)NP01 * 4);
    float*  bs2   = (float*) alloc((size_t)NP2  * 4);
    float*  xg    = (float*) alloc(M * NP01 * 4);        // reused per layer
    __bf16* y0b   = (__bf16*)alloc(M * KP1 * 2);
    __bf16* y1b   = (__bf16*)alloc(M * KP1 * 2);
    float*  gbuf  = (float*) alloc((size_t)128 * NP01 * 4);
    __bf16* hb    = (__bf16*)alloc((size_t)128 * KP1 * 2);
    float*  cst   = (float*) alloc((size_t)128 * KP1 * 4);
    float*  hf    = (float*) alloc((size_t)128 * KP1 * 4);

    auto blks = [](size_t n) { return (unsigned)((n + 255) / 256); };

    // --- one-time (per launch) conversions ---
    cvt_weight<<<blks((size_t)NP01 * KP0), 256, 0, stream>>>(Wih0, wih0b, 4600, NINP, NP01, KP0);
    cvt_weight<<<blks((size_t)NP01 * KP1), 256, 0, stream>>>(Whh0, whh0b, 4600, NHID, NP01, KP1);
    cvt_weight<<<blks((size_t)NP01 * KP1), 256, 0, stream>>>(Wih1, wih1b, 4600, NHID, NP01, KP1);
    cvt_weight<<<blks((size_t)NP01 * KP1), 256, 0, stream>>>(Whh1, whh1b, 4600, NHID, NP01, KP1);
    cvt_weight<<<blks((size_t)NP2  * KP1), 256, 0, stream>>>(Wih2, wih2b, 1600, NHID, NP2, KP1);
    cvt_weight<<<blks((size_t)NP2  * KP0), 256, 0, stream>>>(Whh2, whh2b, 1600, NINP, NP2, KP0);
    bias_sum<<<blks(NP01), 256, 0, stream>>>(bih0, bhh0, bs0, 4600, NP01);
    bias_sum<<<blks(NP01), 256, 0, stream>>>(bih1, bhh1, bs1, 4600, NP01);
    bias_sum<<<blks(NP2),  256, 0, stream>>>(bih2, bhh2, bs2, 1600, NP2);
    embed_kernel<<<blks(M * KP0), 256, 0, stream>>>(tokens, embW, embb, KP0);
    zero_bf16<<<blks(M * KP1), 256, 0, stream>>>(y0b, M * KP1); // pad cols must be 0
    zero_bf16<<<blks(M * KP1), 256, 0, stream>>>(y1b, M * KP1);

    // output offsets (floats)
    size_t o_res = 0;
    size_t o_h0 = M * NINP;                 // 10,240,000
    size_t o_c0 = o_h0 + (size_t)128 * NHID;
    size_t o_h1 = o_c0 + (size_t)128 * NHID;
    size_t o_c1 = o_h1 + (size_t)128 * NHID;
    size_t o_h2 = o_c1 + (size_t)128 * NHID;
    size_t o_c2 = o_h2 + (size_t)128 * NINP;

    // ---------------- layer 0: 400 -> 1150 ----------------
    init_state<<<blks((size_t)128 * KP1), 256, 0, stream>>>(h0_0, c0_0, hb, cst, NHID, KP1);
    wmma_gemm_bias<<<dim3(NP01 / 64, (unsigned)(M / 128)), 256, 0, stream>>>(
        embb, wih0b, bs0, xg, KP0, NP01);
    for (int t = 0; t < 200; ++t) {
        wmma_step_gemm<<<NP01 / 64, 256, 0, stream>>>(
            hb, whh0b, xg + (size_t)t * 128 * NP01, gbuf, KP1, NP01);
        lstm_gates<<<blks((size_t)128 * NHID), 256, 0, stream>>>(
            gbuf, cst, hf, hb, y0b, nullptr, NHID, NP01, KP1, t);
    }
    copy_f32<<<blks((size_t)128 * NHID), 256, 0, stream>>>(out + o_h0, hf, (size_t)128 * NHID);
    copy_f32<<<blks((size_t)128 * NHID), 256, 0, stream>>>(out + o_c0, cst, (size_t)128 * NHID);

    // ---------------- layer 1: 1150 -> 1150 ----------------
    init_state<<<blks((size_t)128 * KP1), 256, 0, stream>>>(h0_1, c0_1, hb, cst, NHID, KP1);
    wmma_gemm_bias<<<dim3(NP01 / 64, (unsigned)(M / 128)), 256, 0, stream>>>(
        y0b, wih1b, bs1, xg, KP1, NP01);
    for (int t = 0; t < 200; ++t) {
        wmma_step_gemm<<<NP01 / 64, 256, 0, stream>>>(
            hb, whh1b, xg + (size_t)t * 128 * NP01, gbuf, KP1, NP01);
        lstm_gates<<<blks((size_t)128 * NHID), 256, 0, stream>>>(
            gbuf, cst, hf, hb, y1b, nullptr, NHID, NP01, KP1, t);
    }
    copy_f32<<<blks((size_t)128 * NHID), 256, 0, stream>>>(out + o_h1, hf, (size_t)128 * NHID);
    copy_f32<<<blks((size_t)128 * NHID), 256, 0, stream>>>(out + o_c1, cst, (size_t)128 * NHID);

    // ---------------- layer 2: 1150 -> 400 (result written to out) ----------------
    init_state<<<blks((size_t)128 * KP0), 256, 0, stream>>>(h0_2, c0_2, hb, cst, NINP, KP0);
    wmma_gemm_bias<<<dim3(NP2 / 64, (unsigned)(M / 128)), 256, 0, stream>>>(
        y1b, wih2b, bs2, xg, KP1, NP2);
    for (int t = 0; t < 200; ++t) {
        wmma_step_gemm<<<NP2 / 64, 256, 0, stream>>>(
            hb, whh2b, xg + (size_t)t * 128 * NP2, gbuf, KP0, NP2);
        lstm_gates<<<blks((size_t)128 * NINP), 256, 0, stream>>>(
            gbuf, cst, hf, hb, nullptr, out + o_res, NINP, NP2, KP0, t);
    }
    copy_f32<<<blks((size_t)128 * NINP), 256, 0, stream>>>(out + o_h2, hf, (size_t)128 * NINP);
    copy_f32<<<blks((size_t)128 * NINP), 256, 0, stream>>>(out + o_c2, cst, (size_t)128 * NINP);
}
